// CausalSelfAttention_58961311039581
// MI455X (gfx1250) — matrix operand order repaired and checked by reading, not compiled, not run on hardware
//
#include <hip/hip_runtime.h>
#include <hip/hip_bf16.h>

typedef __attribute__((ext_vector_type(16))) _Float16 v16h;
typedef __attribute__((ext_vector_type(8)))  _Float16 h8;
typedef __attribute__((ext_vector_type(8)))  float    v8f;

#define D_MODEL   768
#define N_HEADS   12
#define HEAD_DIM  64
#define SEQ       1024
#define BATCH     8
#define ROWS      (BATCH*SEQ)     /* 8192 */
#define QKV_N     (3*D_MODEL)     /* 2304 */

union FragH { v16h v; h8 h[2]; };

__device__ __forceinline__ v8f zero8() {
    v8f z = {0.f,0.f,0.f,0.f,0.f,0.f,0.f,0.f};
    return z;
}

// A-matrix fragment (16x32 f16, MxK). rowk points at element [row][k0] of a
// row-major matrix; lane holds M = lane&15. Halves 0..7 -> K = koff+0..7,
// halves 8..15 -> K = koff+16..23, koff = (lane>=16)?8:0.  (ISA 7.12.2)
__device__ __forceinline__ v16h frag_a_row(const _Float16* rowk, int lane) {
    const int koff = (lane >> 4) * 8;
    FragH f;
    f.h[0] = *(const h8*)(rowk + koff);
    f.h[1] = *(const h8*)(rowk + koff + 16);
    return f.v;
}

// B-matrix fragment (32x16 f16, KxN) from a [N][K]-contiguous source.
// rowk points at element [n][k0]; lane holds N = lane&15, halves 0..7 ->
// K = koff+0..7, halves 8..15 -> K = koff+16..23, koff = (lane>=16)?8:0.
__device__ __forceinline__ v16h frag_b_row(const _Float16* rowk, int lane) {
    const int koff = (lane >> 4) * 8;
    FragH f;
    f.h[0] = *(const h8*)(rowk + koff);
    f.h[1] = *(const h8*)(rowk + koff + 16);
    return f.v;
}

#define WMMA_F16(a,b,c) \
    __builtin_amdgcn_wmma_f32_16x16x32_f16(false,(a),false,(b),(short)0,(c),false,false)

// ---------------------------------------------------------------- cast / transpose
__global__ void cast_f32_to_f16(const float* __restrict__ in,
                                _Float16* __restrict__ out, int n) {
    int i = blockIdx.x * 256 + threadIdx.x;
    if (i < n) out[i] = (_Float16)in[i];
}

// Wt[n*K + k] = W[k*N + n]  (cast to f16)
__global__ void transpose_cast(const float* __restrict__ W,
                               _Float16* __restrict__ Wt, int K, int N) {
    int i = blockIdx.x * 256 + threadIdx.x;
    if (i < K * N) {
        int n = i / K;
        int k = i - n * K;
        Wt[i] = (_Float16)W[(size_t)k * N + n];
    }
}

// ---------------------------------------------------------------- QKV GEMM
// C[8192 x 2304] = Xh[8192 x 768] @ Wqkv, B given transposed [2304 x 768].
// Each wave: 16 rows x 64 cols. Writes q/k/v in [B,H,T,Dh] f16; q scaled.
__global__ void gemm_qkv(const _Float16* __restrict__ A,
                         const _Float16* __restrict__ Bt,
                         _Float16* __restrict__ qh,
                         _Float16* __restrict__ kh,
                         _Float16* __restrict__ vh) {
    const int lane = threadIdx.x & 31;
    const int wave = threadIdx.x >> 5;
    const int task = blockIdx.x * 8 + wave;          // 512*36 = 18432 tasks
    const int NCG  = QKV_N / 64;                     // 36
    const int tm = task / NCG;
    const int tg = task - tm * NCG;
    const int rowbase = tm * 16;
    const int colbase = tg * 64;
    const int ln = lane & 15, hi = lane >> 4;

    v8f acc[4] = { zero8(), zero8(), zero8(), zero8() };

    const _Float16* arow = A + (size_t)(rowbase + ln) * D_MODEL;
    #pragma unroll 4
    for (int k0 = 0; k0 < D_MODEL; k0 += 32) {
        __builtin_prefetch(arow + k0 + 128, 0, 1);
        v16h af = frag_a_row(arow + k0, lane);
        #pragma unroll
        for (int nt = 0; nt < 4; ++nt) {
            const _Float16* brow =
                Bt + (size_t)(colbase + nt * 16 + ln) * D_MODEL + k0;
            v16h bf = frag_b_row(brow, lane);
            acc[nt] = WMMA_F16(af, bf, acc[nt]);
        }
    }

    #pragma unroll
    for (int nt = 0; nt < 4; ++nt) {
        #pragma unroll
        for (int r = 0; r < 8; ++r) {
            int m = rowbase + r + 8 * hi;            // global row (b*T + t)
            int n = colbase + nt * 16 + ln;          // global col in [0,2304)
            int which = n / D_MODEL;                 // 0=q 1=k 2=v (uniform)
            int c = n - which * D_MODEL;
            int h = c >> 6, d = c & 63;
            int b = m >> 10, t = m & 1023;
            float val = acc[nt][r];
            if (which == 0) val *= 0.125f;           // 1/sqrt(64)
            _Float16* dst = (which == 0) ? qh : (which == 1) ? kh : vh;
            dst[((((size_t)b * N_HEADS + h) << 10) + t) * HEAD_DIM + d] =
                (_Float16)val;
        }
    }
}

// ---------------------------------------------------------------- flash attention
// One wave per (bh, 16-query tile). grid = (64, 96), block = 32.
__global__ void attn_flash(const _Float16* __restrict__ qh,
                           const _Float16* __restrict__ kh,
                           const _Float16* __restrict__ vh,
                           _Float16* __restrict__ yh) {
    const int lane = threadIdx.x & 31;
    const int qt = blockIdx.x;                       // 0..63
    const int bh = blockIdx.y;                       // 0..95
    const int b = bh / N_HEADS, h = bh - b * N_HEADS;
    const int ln = lane & 15, hi = lane >> 4;
    const int qbase = qt * 16;

    const _Float16* Q = qh + (size_t)bh * SEQ * HEAD_DIM;
    const _Float16* K = kh + (size_t)bh * SEQ * HEAD_DIM;
    const _Float16* V = vh + (size_t)bh * SEQ * HEAD_DIM;

    __shared__ alignas(16) _Float16 ldsP[16 * 32];      // P tile, C->A relayout
    __shared__ alignas(16) _Float16 ldsVt[HEAD_DIM * 32]; // V^T tile [d][key]

    // Q fragments (16 rows x 64 dims = two 16x32 A-frags), live in registers
    const _Float16* qrow = Q + (size_t)(qbase + ln) * HEAD_DIM;
    v16h qf0 = frag_a_row(qrow + 0, lane);
    v16h qf1 = frag_a_row(qrow + 32, lane);

    float m_i[8], l_i[8];
    v8f o[4] = { zero8(), zero8(), zero8(), zero8() };
    #pragma unroll
    for (int r = 0; r < 8; ++r) { m_i[r] = -1e30f; l_i[r] = 0.f; }

    const int ktmax = (qbase + 15) / 32;             // inclusive
    for (int kt = 0; kt <= ktmax; ++kt) {
        const int kbase = kt * 32;

        // ---- scores S[16x32] as two 16x16 C-frags
        v8f s[2];
        #pragma unroll
        for (int half = 0; half < 2; ++half) {
            const _Float16* krow =
                K + (size_t)(kbase + half * 16 + ln) * HEAD_DIM;
            v16h kf0 = frag_b_row(krow + 0, lane);
            v16h kf1 = frag_b_row(krow + 32, lane);
            v8f c = zero8();
            c = WMMA_F16(qf0, kf0, c);
            c = WMMA_F16(qf1, kf1, c);
            s[half] = c;
        }

        // ---- stage V tile transposed into LDS: ldsVt[d][key]
        {
            const _Float16* vrow = V + (size_t)(kbase + lane) * HEAD_DIM;
            #pragma unroll
            for (int dv = 0; dv < HEAD_DIM; dv += 8) {
                h8 vv = *(const h8*)(vrow + dv);
                #pragma unroll
                for (int j = 0; j < 8; ++j)
                    ldsVt[(dv + j) * 32 + lane] = vv[j];
            }
        }

        // ---- causal mask + online softmax (row = r + 8*hi within tile)
        float alpha[8];
        #pragma unroll
        for (int r = 0; r < 8; ++r) {
            const int qrowg = qbase + r + 8 * hi;
            float s0 = s[0][r], s1 = s[1][r];
            if (kbase + ln > qrowg)      s0 = -1e30f;
            if (kbase + 16 + ln > qrowg) s1 = -1e30f;
            float mx = fmaxf(s0, s1);
            #pragma unroll
            for (int off = 8; off > 0; off >>= 1)
                mx = fmaxf(mx, __shfl_xor(mx, off, 32));
            const float mnew = fmaxf(m_i[r], mx);
            const float a = __expf(m_i[r] - mnew);
            const float p0 = __expf(s0 - mnew);
            const float p1 = __expf(s1 - mnew);
            float ps = p0 + p1;
            #pragma unroll
            for (int off = 8; off > 0; off >>= 1)
                ps += __shfl_xor(ps, off, 32);
            l_i[r] = l_i[r] * a + ps;
            m_i[r] = mnew;
            alpha[r] = a;
            const int mrow = r + 8 * hi;
            ldsP[mrow * 32 + ln]      = (_Float16)p0;
            ldsP[mrow * 32 + 16 + ln] = (_Float16)p1;
        }
        __syncthreads();   // single-wave WG: cheap; orders LDS for compiler

        // ---- rescale accumulators
        #pragma unroll
        for (int nt = 0; nt < 4; ++nt)
            #pragma unroll
            for (int r = 0; r < 8; ++r)
                o[nt][r] *= alpha[r];

        // ---- P A-frag from LDS (rows m = ln, K = key-in-tile)
        v16h pf;
        {
            const int koff = hi * 8;
            FragH f;
            f.h[0] = *(const h8*)(&ldsP[ln * 32 + koff]);
            f.h[1] = *(const h8*)(&ldsP[ln * 32 + koff + 16]);
            pf = f.v;
        }
        // ---- O += P @ V : B-frags contiguous from ldsVt
        #pragma unroll
        for (int nt = 0; nt < 4; ++nt) {
            const int d = nt * 16 + ln;
            const int koff = hi * 8;
            FragH f;
            f.h[0] = *(const h8*)(&ldsVt[d * 32 + koff]);
            f.h[1] = *(const h8*)(&ldsVt[d * 32 + koff + 16]);
            o[nt] = WMMA_F16(pf, f.v, o[nt]);
        }
        __syncthreads();
    }

    // ---- normalize + write y as f16 in [B, T, C] with c = h*64 + d
    #pragma unroll
    for (int nt = 0; nt < 4; ++nt)
        #pragma unroll
        for (int r = 0; r < 8; ++r) {
            const int t = qbase + r + 8 * hi;
            const float val = o[nt][r] / l_i[r];
            yh[((size_t)b * SEQ + t) * D_MODEL + h * HEAD_DIM + nt * 16 + ln] =
                (_Float16)val;
        }
}

// ---------------------------------------------------------------- output projection
// out[8192 x 768] (fp32) = Yh[8192 x 768] @ Wproj (Bt = [768 x 768] transposed)
__global__ void gemm_proj(const _Float16* __restrict__ A,
                          const _Float16* __restrict__ Bt,
                          float* __restrict__ out) {
    const int lane = threadIdx.x & 31;
    const int wave = threadIdx.x >> 5;
    const int task = blockIdx.x * 8 + wave;          // 512*12 = 6144 tasks
    const int NCG  = D_MODEL / 64;                   // 12
    const int tm = task / NCG;
    const int tg = task - tm * NCG;
    const int rowbase = tm * 16;
    const int colbase = tg * 64;
    const int ln = lane & 15, hi = lane >> 4;

    v8f acc[4] = { zero8(), zero8(), zero8(), zero8() };

    const _Float16* arow = A + (size_t)(rowbase + ln) * D_MODEL;
    #pragma unroll 4
    for (int k0 = 0; k0 < D_MODEL; k0 += 32) {
        __builtin_prefetch(arow + k0 + 128, 0, 1);
        v16h af = frag_a_row(arow + k0, lane);
        #pragma unroll
        for (int nt = 0; nt < 4; ++nt) {
            const _Float16* brow =
                Bt + (size_t)(colbase + nt * 16 + ln) * D_MODEL + k0;
            v16h bf = frag_b_row(brow, lane);
            acc[nt] = WMMA_F16(af, bf, acc[nt]);
        }
    }

    #pragma unroll
    for (int nt = 0; nt < 4; ++nt)
        #pragma unroll
        for (int r = 0; r < 8; ++r) {
            const int m = rowbase + r + 8 * hi;
            const int n = colbase + nt * 16 + ln;
            out[(size_t)m * D_MODEL + n] = acc[nt][r];
        }
}

// ---------------------------------------------------------------- launcher
extern "C" void kernel_launch(void* const* d_in, const int* in_sizes, int n_in,
                              void* d_out, int out_size, void* d_ws, size_t ws_size,
                              hipStream_t stream) {
    const float* x      = (const float*)d_in[0];   // [8,1024,768]
    const float* W_qkv  = (const float*)d_in[1];   // [768,2304]
    const float* W_proj = (const float*)d_in[2];   // [768,768]
    float* out = (float*)d_out;                    // [8,1024,768]

    _Float16* ws     = (_Float16*)d_ws;
    _Float16* xh     = ws;                         // 8192*768
    _Float16* wqkvt  = xh + (size_t)ROWS * D_MODEL;        // 2304*768
    _Float16* wprojt = wqkvt + (size_t)QKV_N * D_MODEL;    // 768*768
    _Float16* qh     = wprojt + (size_t)D_MODEL * D_MODEL; // 8*12*1024*64
    _Float16* kh     = qh + (size_t)ROWS * D_MODEL;
    _Float16* vh     = kh + (size_t)ROWS * D_MODEL;
    _Float16* yh     = vh + (size_t)ROWS * D_MODEL;

    const int nx = ROWS * D_MODEL;                 // 6291456
    cast_f32_to_f16<<<(nx + 255) / 256, 256, 0, stream>>>(x, xh, nx);

    const int nwq = D_MODEL * QKV_N;               // 1769472
    transpose_cast<<<(nwq + 255) / 256, 256, 0, stream>>>(W_qkv, wqkvt,
                                                          D_MODEL, QKV_N);
    const int nwp = D_MODEL * D_MODEL;             // 589824
    transpose_cast<<<(nwp + 255) / 256, 256, 0, stream>>>(W_proj, wprojt,
                                                          D_MODEL, D_MODEL);

    // 18432 wave-tasks, 8 waves per 256-thread block
    gemm_qkv<<<(ROWS / 16) * (QKV_N / 64) / 8, 256, 0, stream>>>(
        xh, wqkvt, qh, kh, vh);

    dim3 agrid(SEQ / 16, BATCH * N_HEADS);         // (64, 96)
    attn_flash<<<agrid, 32, 0, stream>>>(qh, kh, vh, yh);

    gemm_proj<<<(ROWS / 16) * (D_MODEL / 64) / 8, 256, 0, stream>>>(
        yh, wprojt, out);
}
